// GraphSAGE_86577950753302
// MI455X (gfx1250) — compile-verified, matrix operand order checked by reading, not run on hardware
//
#include <hip/hip_runtime.h>
#include <hip/hip_bf16.h>

typedef __attribute__((ext_vector_type(16))) __bf16 v16bf;
typedef __attribute__((ext_vector_type(8)))  __bf16 v8bf;
typedef __attribute__((ext_vector_type(8)))  float  v8f;

#define N_NODES   100000
#define N_EDGES   1600000
#define IN_DIM    128
#define HIDDEN    128
#define N_CLASSES 40

// ---------------------------------------------------------------------------
// Degree histogram: deg[dst[e]] += 1
// ---------------------------------------------------------------------------
__global__ __launch_bounds__(256) void deg_kernel(const int* __restrict__ dst,
                                                  float* __restrict__ deg,
                                                  int n_edges) {
    int e = blockIdx.x * blockDim.x + threadIdx.x;
    if (e < n_edges) atomicAdd(&deg[dst[e]], 1.0f);
}

// ---------------------------------------------------------------------------
// Pack [Wl; Wr] (K=256 concat) into column-major bf16, zero-padded columns.
// wpack[col*256 + k]  (k<128 -> Wl[k][col], k>=128 -> Wr[k-128][col])
// ---------------------------------------------------------------------------
__global__ __launch_bounds__(256) void pack_w_kernel(const float* __restrict__ Wl,
                                                     const float* __restrict__ Wr,
                                                     __bf16* __restrict__ wpack,
                                                     int out_dim, int padded) {
    int idx = blockIdx.x * blockDim.x + threadIdx.x;
    if (idx >= padded * 256) return;
    int colp = idx >> 8;
    int k    = idx & 255;
    float v = 0.0f;
    if (colp < out_dim)
        v = (k < 128) ? Wl[k * out_dim + colp] : Wr[(k - 128) * out_dim + colp];
    wpack[idx] = (__bf16)v;
}

// ---------------------------------------------------------------------------
// Edge scatter: ssum[dst] += feat[src].  One wave per edge, lane owns 4 floats.
// Node features (51.2MB) + accumulator (51.2MB) both fit in the 192MB L2,
// so the random gather/atomic traffic stays on-chip.
// ---------------------------------------------------------------------------
__global__ __launch_bounds__(256) void scatter_kernel(const float* __restrict__ feat,
                                                      const int* __restrict__ src,
                                                      const int* __restrict__ dst,
                                                      float* __restrict__ ssum,
                                                      int n_edges) {
    const int lane = threadIdx.x & 31;
    const int e = blockIdx.x * 8 + (threadIdx.x >> 5);
    if (e >= n_edges) return;
    const int s = src[e];
    const int d = dst[e];
    const float4 v = *(const float4*)(feat + (size_t)s * 128 + lane * 4);
    float* o = ssum + (size_t)d * 128 + lane * 4;
    atomicAdd(o + 0, v.x);
    atomicAdd(o + 1, v.y);
    atomicAdd(o + 2, v.z);
    atomicAdd(o + 3, v.w);
}

// ---------------------------------------------------------------------------
// Finalize: materialize the concatenated GEMM operand in bf16, once per layer.
//   acat[row][k] = bf16( ssum[row][k] / max(deg,1) )   for k < 128   (mean)
//   acat[row][k] = bf16( self[row][k-128] )            for k >= 128  (self)
// Thread handles 4 consecutive features -> float4 load, 2x cvt_pk, 8B store.
// ---------------------------------------------------------------------------
__global__ __launch_bounds__(256) void finalize_kernel(const float* __restrict__ ssum,
                                                       const float* __restrict__ deg,
                                                       const float* __restrict__ self,
                                                       __bf16* __restrict__ acat) {
    const long idx = (long)blockIdx.x * blockDim.x + threadIdx.x;  // over N*64
    if (idx >= (long)N_NODES * 64) return;
    const int row = (int)(idx >> 6);
    const int k4  = ((int)idx & 63) * 4;     // 0..252

    float4 f;
    float scale;
    if (k4 < 128) {
        f = *(const float4*)(ssum + (size_t)row * 128 + k4);
        scale = 1.0f / fmaxf(deg[row], 1.0f);
    } else {
        f = *(const float4*)(self + (size_t)row * 128 + (k4 - 128));
        scale = 1.0f;
    }
    __bf16 o[4];
    o[0] = (__bf16)(f.x * scale);
    o[1] = (__bf16)(f.y * scale);
    o[2] = (__bf16)(f.z * scale);
    o[3] = (__bf16)(f.w * scale);
    *(ulong1*)(acat + (size_t)row * 256 + k4) = *(ulong1*)o;   // 8-byte store
}

// ---------------------------------------------------------------------------
// SAGE layer GEMM: out = relu?( acat @ Wpack + b ),  acat bf16 [N][256].
// One wave computes one 16x16 tile via 8x v_wmma_f32_16x16x32_bf16 (K=256).
// Both A and B fragments are pure contiguous 16B bf16 loads per K-span
// (ISA 16-bit A/B layout: lane<16 -> spans [kc*32+0..7],[kc*32+16..23];
//  lane>=16 -> +8), so the inner loop is 4x global_load_b128 + 1x wmma.
// ---------------------------------------------------------------------------
__global__ __launch_bounds__(256) void sage_wmma_kernel(
        const __bf16* __restrict__ acat,
        const __bf16* __restrict__ wpack,
        const float*  __restrict__ bias,
        float*        __restrict__ out,
        int out_dim, int n_ntiles, int relu) {
    const int lane   = threadIdx.x & 31;
    const int mtiles = N_NODES / 16;                 // 6250, exact
    const int tile   = blockIdx.x * 8 + (threadIdx.x >> 5);
    if (tile >= mtiles * n_ntiles) return;           // wave-uniform: EXEC stays all-1s
    const int mt = tile / n_ntiles;
    const int nt = tile - mt * n_ntiles;

    const int row   = mt * 16 + (lane & 15);
    const int khalf = (lane & 16) ? 8 : 0;
    const int col   = nt * 16 + (lane & 15);

    const __bf16* __restrict__ arow = acat  + (size_t)row * 256;
    const __bf16* __restrict__ wcol = wpack + (size_t)col * 256;

    v8f acc = {};
#pragma unroll
    for (int kc = 0; kc < 8; ++kc) {
        const int kbase = kc * 32 + khalf;
        const v8bf a0 = *(const v8bf*)(arow + kbase);        // 16B
        const v8bf a1 = *(const v8bf*)(arow + kbase + 16);   // 16B
        const v8bf b0 = *(const v8bf*)(wcol + kbase);        // 16B
        const v8bf b1 = *(const v8bf*)(wcol + kbase + 16);   // 16B
        v16bf a, b;
#pragma unroll
        for (int i = 0; i < 8; ++i) {
            a[i] = a0[i]; a[i + 8] = a1[i];
            b[i] = b0[i]; b[i + 8] = b1[i];
        }
        acc = __builtin_amdgcn_wmma_f32_16x16x32_bf16(
                  false, a, false, b, (short)0, acc, false, false);
    }

    // D layout: VGPR v -> M = v (+8 for upper half-wave), N = lane&15
    if (col < out_dim) {
        const float bc    = bias[col];
        const int   rbase = mt * 16 + ((lane & 16) ? 8 : 0);
#pragma unroll
        for (int v = 0; v < 8; ++v) {
            float val = acc[v] + bc;
            if (relu) val = fmaxf(val, 0.0f);
            out[(size_t)(rbase + v) * out_dim + col] = val;
        }
    }
}

// ---------------------------------------------------------------------------
extern "C" void kernel_launch(void* const* d_in, const int* in_sizes, int n_in,
                              void* d_out, int out_size, void* d_ws, size_t ws_size,
                              hipStream_t stream) {
    (void)in_sizes; (void)n_in; (void)out_size; (void)ws_size;

    const float* x   = (const float*)d_in[0];
    const int*   ei  = (const int*)d_in[1];
    const float* Wl1 = (const float*)d_in[2];
    const float* Wr1 = (const float*)d_in[3];
    const float* b1  = (const float*)d_in[4];
    const float* Wl2 = (const float*)d_in[5];
    const float* Wr2 = (const float*)d_in[6];
    const float* b2  = (const float*)d_in[7];
    const int* src = ei;
    const int* dst = ei + N_EDGES;

    char* ws = (char*)d_ws;
    size_t off = 0;
    auto take = [&](size_t bytes) {
        char* p = ws + off;
        off = (off + bytes + 255) & ~(size_t)255;
        return p;
    };
    float*  deg  = (float*) take((size_t)N_NODES * 4);
    float*  ssum = (float*) take((size_t)N_NODES * 128 * 4);
    float*  h    = (float*) take((size_t)N_NODES * 128 * 4);
    __bf16* acat = (__bf16*)take((size_t)N_NODES * 256 * 2);
    __bf16* wp1  = (__bf16*)take((size_t)128 * 256 * 2);
    __bf16* wp2  = (__bf16*)take((size_t)48  * 256 * 2);

    hipMemsetAsync(deg,  0, (size_t)N_NODES * 4, stream);
    hipMemsetAsync(ssum, 0, (size_t)N_NODES * 128 * 4, stream);

    deg_kernel<<<(N_EDGES + 255) / 256, 256, 0, stream>>>(dst, deg, N_EDGES);
    pack_w_kernel<<<(128 * 256 + 255) / 256, 256, 0, stream>>>(Wl1, Wr1, wp1, HIDDEN, 128);
    pack_w_kernel<<<(48  * 256 + 255) / 256, 256, 0, stream>>>(Wl2, Wr2, wp2, N_CLASSES, 48);

    const int finBlocks = (int)(((long)N_NODES * 64 + 255) / 256);

    // Layer 1
    scatter_kernel<<<(N_EDGES + 7) / 8, 256, 0, stream>>>(x, src, dst, ssum, N_EDGES);
    finalize_kernel<<<finBlocks, 256, 0, stream>>>(ssum, deg, x, acat);
    {
        const int tiles = (N_NODES / 16) * 8;   // 8 n-tiles (HIDDEN=128)
        sage_wmma_kernel<<<(tiles + 7) / 8, 256, 0, stream>>>(
            acat, wp1, b1, h, HIDDEN, 8, 1);
    }

    // Layer 2
    hipMemsetAsync(ssum, 0, (size_t)N_NODES * 128 * 4, stream);
    scatter_kernel<<<(N_EDGES + 7) / 8, 256, 0, stream>>>(h, src, dst, ssum, N_EDGES);
    finalize_kernel<<<finBlocks, 256, 0, stream>>>(ssum, deg, h, acat);
    {
        const int tiles = (N_NODES / 16) * 3;   // 3 n-tiles (40 classes padded to 48)
        sage_wmma_kernel<<<(tiles + 7) / 8, 256, 0, stream>>>(
            acat, wp2, b2, (float*)d_out, N_CLASSES, 3, 0);
    }
}
